// GPTDecoder_11149735100449
// MI455X (gfx1250) — matrix operand order repaired and checked by reading, not compiled, not run on hardware
//
#include <hip/hip_runtime.h>
#include <hip/hip_bf16.h>

// ---------------------------------------------------------------------------
// GPT-2-ish decoder forward for MI455X (gfx1250, wave32, WMMA bf16).
// V=50257 D=1024 H=16 DFF=4096 L=6 S=2048 B=2
// ---------------------------------------------------------------------------

typedef __bf16 bf16_t;
typedef __attribute__((ext_vector_type(16))) __bf16 v16bf;
typedef __attribute__((ext_vector_type(8)))  __bf16 v8bf;
typedef __attribute__((ext_vector_type(8)))  float  v8f;

__device__ __forceinline__ bf16_t f2bf(float f) {
  unsigned u = __builtin_bit_cast(unsigned, f);
  unsigned r = u + 0x7fffu + ((u >> 16) & 1u);           // round-to-nearest-even
  unsigned short h = (unsigned short)(r >> 16);
  return __builtin_bit_cast(bf16_t, h);
}

__device__ __forceinline__ v8f wmma_bf16(v16bf a, v16bf b, v8f c) {
  // D = A(16x32 bf16) * B(32x16 bf16) + C(16x16 f32)
  return __builtin_amdgcn_wmma_f32_16x16x32_bf16(
      /*neg_a=*/false, a, /*neg_b=*/false, b,
      /*c_mod=*/(short)0, c, /*reuse_a=*/false, /*reuse_b=*/false);
}

// Async global->LDS copy (16B per lane), tracked by ASYNCcnt.
// Generic pointers to __shared__ carry the LDS byte offset in bits [31:0].
__device__ __forceinline__ void async_copy_b128(const bf16_t* g, bf16_t* l) {
  unsigned lds_off = (unsigned)(unsigned long long)l;
  unsigned long long ga = (unsigned long long)g;
  asm volatile("global_load_async_to_lds_b128 %0, %1, off"
               :: "v"(lds_off), "v"(ga) : "memory");
}
__device__ __forceinline__ void wait_async() {
  asm volatile("s_wait_asynccnt 0x0" ::: "memory");
}

// ---------------------------------------------------------------------------
// Embedding gather: x[b,s,:] = tok_emb[id] + pos_emb[s]   (fp32)
// ---------------------------------------------------------------------------
__global__ __launch_bounds__(256) void embed_kernel(
    const int* __restrict__ ids, const float* __restrict__ tok,
    const float* __restrict__ pos, float* __restrict__ x) {
  constexpr int Dm = 1024, Sl = 2048;
  int row = blockIdx.x, d = threadIdx.x * 4;
  int id = ids[row];
  int sp = row % Sl;
  float4 t = *(const float4*)(tok + (size_t)id * Dm + d);
  float4 p = *(const float4*)(pos + (size_t)sp * Dm + d);
  *(float4*)(x + (size_t)row * Dm + d) =
      make_float4(t.x + p.x, t.y + p.y, t.z + p.z, t.w + p.w);
}

// ---------------------------------------------------------------------------
// LayerNorm over D=1024, fp32 in -> bf16 out. One block (8 waves) per row.
// ---------------------------------------------------------------------------
__global__ __launch_bounds__(256) void ln_kernel(
    const float* __restrict__ x, const float* __restrict__ g,
    const float* __restrict__ bt, bf16_t* __restrict__ out) {
  constexpr int Dm = 1024;
  int row = blockIdx.x, tid = threadIdx.x;
  const float* xr = x + (size_t)row * Dm;
  float4 xv = *(const float4*)(xr + tid * 4);
  float s  = xv.x + xv.y + xv.z + xv.w;
  float s2 = xv.x * xv.x + xv.y * xv.y + xv.z * xv.z + xv.w * xv.w;
#pragma unroll
  for (int off = 1; off < 32; off <<= 1) {
    s  += __shfl_xor(s,  off, 32);
    s2 += __shfl_xor(s2, off, 32);
  }
  __shared__ float red[2][8];
  if ((tid & 31) == 0) { red[0][tid >> 5] = s; red[1][tid >> 5] = s2; }
  __syncthreads();
  float S1 = 0.f, S2 = 0.f;
#pragma unroll
  for (int i = 0; i < 8; ++i) { S1 += red[0][i]; S2 += red[1][i]; }
  float mu = S1 * (1.0f / Dm);
  float rstd = rsqrtf(S2 * (1.0f / Dm) - mu * mu + 1e-5f);
  int d = tid * 4;
  float4 gv = *(const float4*)(g + d);
  float4 bv = *(const float4*)(bt + d);
  bf16_t* orow = out + (size_t)row * Dm + d;
  orow[0] = f2bf((xv.x - mu) * rstd * gv.x + bv.x);
  orow[1] = f2bf((xv.y - mu) * rstd * gv.y + bv.y);
  orow[2] = f2bf((xv.z - mu) * rstd * gv.z + bv.z);
  orow[3] = f2bf((xv.w - mu) * rstd * gv.w + bv.w);
}

// ---------------------------------------------------------------------------
// WMMA bf16 GEMM: C[M,N] = epilogue(A[M,K]bf16 @ B[K,N]f32->bf16)
// Block tile 128x128, 8 waves (4m x 2n), 32x64 per wave (8 WMMA / k-step),
// K step 32. A tile staged with async global->LDS copies.
// flags: 1 = out bf16, 2 = residual add (f32 out), 4 = exact GELU,
//        8 = B stored column-major (element (k,n) at B[n*ldb + k])
// ---------------------------------------------------------------------------
#define GF_OUTBF16 1
#define GF_RES     2
#define GF_GELU    4
#define GF_BCOL    8

__global__ __launch_bounds__(256) void gemm_bf16(
    const bf16_t* __restrict__ A, int lda, const float* __restrict__ B, int ldb,
    const float* __restrict__ bias, void* __restrict__ Cout,
    int M, int N, int K, int flags) {
  __shared__ __align__(16) bf16_t As[128][40];   // [m][k] 32-K tile + pad
  __shared__ __align__(16) bf16_t Bt[128][40];   // transposed: [n][k] + pad

  int tid = threadIdx.x, lane = tid & 31, wave = tid >> 5;
  int wm = wave & 3, wn = wave >> 2;
  int hf = lane >> 4, lm = lane & 15;
  int m0 = blockIdx.y * 128, n0 = blockIdx.x * 128;

  const v8f zf = {0.f, 0.f, 0.f, 0.f, 0.f, 0.f, 0.f, 0.f};
  v8f acc[2][4];
#pragma unroll
  for (int i = 0; i < 2; ++i)
#pragma unroll
    for (int j = 0; j < 4; ++j) acc[i][j] = zf;

  for (int k0 = 0; k0 < K; k0 += 32) {
    __syncthreads();
    {  // stage A: 128x32 bf16, 32B per thread via async global->LDS DMA
      int row = tid >> 1, c = (tid & 1) * 16;
      const bf16_t* src = A + (size_t)(m0 + row) * lda + k0 + c;
      async_copy_b128(src,     &As[row][c]);
      async_copy_b128(src + 8, &As[row][c + 8]);
    }
    if (flags & GF_BCOL) {  // stage B (col-major): coalesced along K
      int n = tid >> 1, kq = (tid & 1) * 16;
      bool ok = (n0 + n) < N;
      const float* bp = B + (size_t)(n0 + n) * ldb + k0 + kq;
#pragma unroll
      for (int q = 0; q < 4; ++q) {
        float4 bv = ok ? *(const float4*)(bp + q * 4)
                       : make_float4(0.f, 0.f, 0.f, 0.f);
        Bt[n][kq + q * 4]     = f2bf(bv.x);
        Bt[n][kq + q * 4 + 1] = f2bf(bv.y);
        Bt[n][kq + q * 4 + 2] = f2bf(bv.z);
        Bt[n][kq + q * 4 + 3] = f2bf(bv.w);
      }
    } else {  // stage B (row-major weights): coalesced along N, transpose in LDS
      int kk = tid >> 4, n8 = (tid & 15) * 8;
      if (k0 + 32 < K)  // prefetch next k-tile of B into L2
        __builtin_prefetch(B + (size_t)(k0 + 32 + kk) * ldb + n0 + n8, 0, 1);
#pragma unroll
      for (int h2 = 0; h2 < 2; ++h2) {
        int kr = kk + h2 * 16;
        const float* bp = B + (size_t)(k0 + kr) * ldb + n0 + n8;
        float4 b0 = *(const float4*)bp;
        float4 b1 = *(const float4*)(bp + 4);
        Bt[n8 + 0][kr] = f2bf(b0.x);
        Bt[n8 + 1][kr] = f2bf(b0.y);
        Bt[n8 + 2][kr] = f2bf(b0.z);
        Bt[n8 + 3][kr] = f2bf(b0.w);
        Bt[n8 + 4][kr] = f2bf(b1.x);
        Bt[n8 + 5][kr] = f2bf(b1.y);
        Bt[n8 + 6][kr] = f2bf(b1.z);
        Bt[n8 + 7][kr] = f2bf(b1.w);
      }
    }
    wait_async();
    __syncthreads();

    v16bf afrag[2], bfrag[4];
#pragma unroll
    for (int mt = 0; mt < 2; ++mt) {  // A layout: lane=M, chunks hf*8 / 16+hf*8
      const bf16_t* ap = &As[wm * 32 + mt * 16 + lm][hf * 8];
      ((v8bf*)&afrag[mt])[0] = *(const v8bf*)ap;
      ((v8bf*)&afrag[mt])[1] = *(const v8bf*)(ap + 16);
    }
#pragma unroll
    for (int nt = 0; nt < 4; ++nt) {  // B layout: lane=N, chunks hf*8 / 16+hf*8
      const bf16_t* bp = &Bt[wn * 64 + nt * 16 + lm][hf * 8];
      ((v8bf*)&bfrag[nt])[0] = *(const v8bf*)bp;
      ((v8bf*)&bfrag[nt])[1] = *(const v8bf*)(bp + 16);
    }
#pragma unroll
    for (int mt = 0; mt < 2; ++mt)
#pragma unroll
      for (int nt = 0; nt < 4; ++nt)
        acc[mt][nt] = wmma_bf16(afrag[mt], bfrag[nt], acc[mt][nt]);
  }

  // epilogue (C/D layout: row = r + 8*hf, col = lane&15)
#pragma unroll
  for (int mt = 0; mt < 2; ++mt)
#pragma unroll
    for (int nt = 0; nt < 4; ++nt)
#pragma unroll
      for (int r = 0; r < 8; ++r) {
        int row = m0 + wm * 32 + mt * 16 + r + hf * 8;
        int col = n0 + wn * 64 + nt * 16 + lm;
        if (col < N) {
          float v = acc[mt][nt][r];
          if (bias) v += bias[col];
          if (flags & GF_GELU) v = 0.5f * v * (1.0f + erff(v * 0.70710678f));
          size_t idx = (size_t)row * N + col;
          if (flags & GF_OUTBF16) {
            ((bf16_t*)Cout)[idx] = f2bf(v);
          } else {
            float* Co = (float*)Cout;
            if (flags & GF_RES) v += Co[idx];
            Co[idx] = v;
          }
        }
      }
}

// ---------------------------------------------------------------------------
// Flash attention (causal). grid=(S/64, H, B), 128 threads (4 waves).
// Each wave owns 16 query rows; K/V tiles of 32 keys staged in LDS
// (K tile via async global->LDS DMA; V tile transposed through VGPRs).
// q,k,v,o are bf16 [B,S,D] with head slice at column h*64.
// ---------------------------------------------------------------------------
__global__ __launch_bounds__(128) void attn_kernel(
    const bf16_t* __restrict__ Q, const bf16_t* __restrict__ Kg,
    const bf16_t* __restrict__ Vg, bf16_t* __restrict__ Og) {
  constexpr int S_ = 2048, D_ = 1024;
  __shared__ __align__(16) bf16_t Ks[32][72];    // [key][dk]   (144B rows)
  __shared__ __align__(16) bf16_t Vs[64][40];    // [dk][key]   (80B rows)
  __shared__ __align__(16) float  Ps[4][16][36]; // per-wave P tile (144B rows)

  int qb = blockIdx.x * 64;
  int h  = blockIdx.y, b = blockIdx.z;
  int tid = threadIdx.x, lane = tid & 31, w = tid >> 5;
  int hf = lane >> 4, lm = lane & 15;

  // Q A-fragments for this wave's 16 rows (dk = 64 -> two K=32 chunks)
  const bf16_t* qp = Q + ((size_t)b * S_ + qb + w * 16 + lm) * D_ + h * 64;
  v16bf qa[2];
#pragma unroll
  for (int c = 0; c < 2; ++c) {
    ((v8bf*)&qa[c])[0] = *(const v8bf*)(qp + c * 32 + hf * 8);
    ((v8bf*)&qa[c])[1] = *(const v8bf*)(qp + c * 32 + 16 + hf * 8);
  }

  const v8f zf = {0.f, 0.f, 0.f, 0.f, 0.f, 0.f, 0.f, 0.f};
  v8f o[4];
  float mrow[8], lrow[8];
#pragma unroll
  for (int nb = 0; nb < 4; ++nb) o[nb] = zf;
#pragma unroll
  for (int r = 0; r < 8; ++r) { mrow[r] = -3.0e38f; lrow[r] = 0.f; }

  int wqmax = qb + w * 16 + 15;
  for (int kb = 0; kb < qb + 64; kb += 32) {
    __syncthreads();
    {  // cooperative staging: 128 threads cover 32 keys x 4 dk-chunks
      int key = tid & 31, dk = (tid >> 5) * 16;
      const bf16_t* kp = Kg + ((size_t)b * S_ + kb + key) * D_ + h * 64 + dk;
      async_copy_b128(kp,     &Ks[key][dk]);
      async_copy_b128(kp + 8, &Ks[key][dk + 8]);
      const bf16_t* vp = Vg + ((size_t)b * S_ + kb + key) * D_ + h * 64 + dk;
      v8bf v0 = ((const v8bf*)vp)[0], v1 = ((const v8bf*)vp)[1];
#pragma unroll
      for (int i = 0; i < 8; ++i) Vs[dk + i][key] = v0[i];
#pragma unroll
      for (int i = 0; i < 8; ++i) Vs[dk + 8 + i][key] = v1[i];
    }
    wait_async();
    __syncthreads();

    if (kb <= wqmax) {
      // scores S = Q @ K^T : two 16x16 D tiles (keys t*16..t*16+15)
      v8f s[2];
      s[0] = zf; s[1] = zf;
#pragma unroll
      for (int t = 0; t < 2; ++t)
#pragma unroll
        for (int c = 0; c < 2; ++c) {
          v16bf kf;
          const bf16_t* kp = &Ks[t * 16 + lm][c * 32 + hf * 8];
          ((v8bf*)&kf)[0] = *(const v8bf*)kp;
          ((v8bf*)&kf)[1] = *(const v8bf*)(kp + 16);
          s[t] = wmma_bf16(qa[c], kf, s[t]);
        }
      // scale + causal mask
#pragma unroll
      for (int t = 0; t < 2; ++t)
#pragma unroll
        for (int r = 0; r < 8; ++r) {
          int qrow = qb + w * 16 + r + hf * 8;
          int key  = kb + t * 16 + lm;
          float sv = s[t][r] * 0.125f;  // 1/sqrt(64)
          s[t][r] = (key <= qrow) ? sv : -3.0e38f;
        }
      // online softmax (row-wise across the 16-lane N group)
#pragma unroll
      for (int r = 0; r < 8; ++r) {
        float rm = fmaxf(s[0][r], s[1][r]);
#pragma unroll
        for (int off = 1; off < 16; off <<= 1)
          rm = fmaxf(rm, __shfl_xor(rm, off, 32));
        float mnew  = fmaxf(mrow[r], rm);
        float alpha = __expf(mrow[r] - mnew);
        mrow[r] = mnew;
        float p0 = __expf(s[0][r] - mnew);
        float p1 = __expf(s[1][r] - mnew);
        s[0][r] = p0; s[1][r] = p1;
        float rs = p0 + p1;
#pragma unroll
        for (int off = 1; off < 16; off <<= 1) rs += __shfl_xor(rs, off, 32);
        lrow[r] = lrow[r] * alpha + rs;
#pragma unroll
        for (int nb = 0; nb < 4; ++nb) o[nb][r] *= alpha;
      }
      // reshape P: D-layout f32 -> A-layout bf16 via wave-private LDS
#pragma unroll
      for (int t = 0; t < 2; ++t)
#pragma unroll
        for (int r = 0; r < 8; ++r)
          Ps[w][r + hf * 8][t * 16 + lm] = s[t][r];
      v16bf pa;
      {
        v8f c0 = *(const v8f*)&Ps[w][lm][hf * 8];
        v8f c1 = *(const v8f*)&Ps[w][lm][16 + hf * 8];
#pragma unroll
        for (int e = 0; e < 8; ++e) { pa[e] = f2bf(c0[e]); pa[e + 8] = f2bf(c1[e]); }
      }
      // O += P(16x32) @ V(32x64)
#pragma unroll
      for (int nb = 0; nb < 4; ++nb) {
        v16bf vf;
        const bf16_t* vp2 = &Vs[nb * 16 + lm][hf * 8];
        ((v8bf*)&vf)[0] = *(const v8bf*)vp2;
        ((v8bf*)&vf)[1] = *(const v8bf*)(vp2 + 16);
        o[nb] = wmma_bf16(pa, vf, o[nb]);
      }
    }
  }

  // normalize + write bf16 out at [b, q, h*64 + dk]
#pragma unroll
  for (int nb = 0; nb < 4; ++nb)
#pragma unroll
    for (int r = 0; r < 8; ++r) {
      int qrow = qb + w * 16 + r + hf * 8;
      int col  = h * 64 + nb * 16 + lm;
      Og[((size_t)b * S_ + qrow) * D_ + col] = f2bf(o[nb][r] / lrow[r]);
    }
}

// ---------------------------------------------------------------------------
// Host-side orchestration
// ---------------------------------------------------------------------------
extern "C" void kernel_launch(void* const* d_in, const int* in_sizes, int n_in,
                              void* d_out, int out_size, void* d_ws, size_t ws_size,
                              hipStream_t stream) {
  constexpr int Vv = 50257, Dd = 1024, Hh = 16, Ff = 4096, Ll = 6, Ss = 2048, Bb = 2;
  constexpr int Mr = Bb * Ss;  // 4096 token rows

  const int*   ids  = (const int*)d_in[0];
  const float* tok  = (const float*)d_in[1];
  const float* pos  = (const float*)d_in[2];
  const float* wq   = (const float*)d_in[3];
  const float* wk   = (const float*)d_in[4];
  const float* wv   = (const float*)d_in[5];
  const float* wo   = (const float*)d_in[6];
  const float* bo   = (const float*)d_in[7];
  const float* ln1g = (const float*)d_in[8];
  const float* ln1b = (const float*)d_in[9];
  const float* ln2g = (const float*)d_in[10];
  const float* ln2b = (const float*)d_in[11];
  const float* w1   = (const float*)d_in[12];
  const float* b1   = (const float*)d_in[13];
  const float* w2   = (const float*)d_in[14];
  const float* b2   = (const float*)d_in[15];
  const float* lnfg = (const float*)d_in[16];
  const float* lnfb = (const float*)d_in[17];

  // workspace carve-up
  char* ws = (char*)d_ws;
  size_t off = 0;
  float*  x    = (float*)(ws + off);  off += (size_t)Mr * Dd * sizeof(float);
  bf16_t* hbuf = (bf16_t*)(ws + off); off += (size_t)Mr * Dd * sizeof(bf16_t);
  bf16_t* qb   = (bf16_t*)(ws + off); off += (size_t)Mr * Dd * sizeof(bf16_t);
  bf16_t* kb   = (bf16_t*)(ws + off); off += (size_t)Mr * Dd * sizeof(bf16_t);
  bf16_t* vb   = (bf16_t*)(ws + off); off += (size_t)Mr * Dd * sizeof(bf16_t);
  bf16_t* ab   = (bf16_t*)(ws + off); off += (size_t)Mr * Dd * sizeof(bf16_t);
  bf16_t* mid  = (bf16_t*)(ws + off); off += (size_t)Mr * Ff * sizeof(bf16_t);
  (void)ws_size; (void)in_sizes; (void)n_in; (void)out_size;

  embed_kernel<<<Mr, 256, 0, stream>>>(ids, tok, pos, x);

  dim3 gD(Dd / 128, Mr / 128);   // N=1024 GEMMs
  dim3 gF(Ff / 128, Mr / 128);   // N=4096 GEMM
  dim3 gA(Ss / 64, Hh, Bb);      // attention

  for (int l = 0; l < Ll; ++l) {
    ln_kernel<<<Mr, 256, 0, stream>>>(x, ln1g + (size_t)l * Dd, ln1b + (size_t)l * Dd, hbuf);
    gemm_bf16<<<gD, 256, 0, stream>>>(hbuf, Dd, wq + (size_t)l * Dd * Dd, Dd,
                                      nullptr, qb, Mr, Dd, Dd, GF_OUTBF16);
    gemm_bf16<<<gD, 256, 0, stream>>>(hbuf, Dd, wk + (size_t)l * Dd * Dd, Dd,
                                      nullptr, kb, Mr, Dd, Dd, GF_OUTBF16);
    gemm_bf16<<<gD, 256, 0, stream>>>(hbuf, Dd, wv + (size_t)l * Dd * Dd, Dd,
                                      nullptr, vb, Mr, Dd, Dd, GF_OUTBF16);
    attn_kernel<<<gA, 128, 0, stream>>>(qb, kb, vb, ab);
    gemm_bf16<<<gD, 256, 0, stream>>>(ab, Dd, wo + (size_t)l * Dd * Dd, Dd,
                                      bo + (size_t)l * Dd, x, Mr, Dd, Dd, GF_RES);
    ln_kernel<<<Mr, 256, 0, stream>>>(x, ln2g + (size_t)l * Dd, ln2b + (size_t)l * Dd, hbuf);
    gemm_bf16<<<gF, 256, 0, stream>>>(hbuf, Dd, w1 + (size_t)l * Dd * Ff, Ff,
                                      b1 + (size_t)l * Ff, mid, Mr, Ff, Dd,
                                      GF_OUTBF16 | GF_GELU);
    gemm_bf16<<<gD, 256, 0, stream>>>(mid, Ff, w2 + (size_t)l * Ff * Dd, Dd,
                                      b2 + (size_t)l * Dd, x, Mr, Dd, Ff, GF_RES);
  }

  ln_kernel<<<Mr, 256, 0, stream>>>(x, lnfg, lnfb, hbuf);
  // tied LM head: logits = h @ tok_emb^T  (B column-major, ragged N=50257)
  gemm_bf16<<<dim3((Vv + 127) / 128, Mr / 128), 256, 0, stream>>>(
      hbuf, Dd, tok, Dd, nullptr, d_out, Mr, Vv, Dd, GF_BCOL);
}
